// PoseSelective_P4CNN_MNIST_12592844112126
// MI455X (gfx1250) — compile-verified, matrix operand order checked by reading, not run on hardware
//
#include <hip/hip_runtime.h>
#include <hip/hip_bf16.h>

typedef _Float16 f16;
typedef __attribute__((ext_vector_type(8)))  _Float16 v8h;
typedef __attribute__((ext_vector_type(16))) _Float16 v16h;
typedef __attribute__((ext_vector_type(8)))  float    v8f;

#define NCH  40     // C*G channels
#define NPAD 48     // padded N for 3x 16-wide WMMA tiles
#define EPSV 1e-5f

// ---------------- small utility kernels ----------------

__global__ void zero_f32(float* p, int n) {
  int i = blockIdx.x * blockDim.x + threadIdx.x;
  if (i < n) p[i] = 0.f;
}

__global__ void cvt_f32_f16(const float* __restrict__ in, f16* __restrict__ out, long n) {
  long i = (long)blockIdx.x * blockDim.x + threadIdx.x;
  if (i < n) out[i] = (f16)in[i];
}

// ---------------- weight expansion (group equivariance) ----------------
// Lifted P4 kernel: full[c*4+r, 0, dy, dx] = rot90^r(w1[c,0])[dy,dx]
// Weight matrix layout: [Kpad=32][48] f16, k = dy*3+dx (Ci=1), n = c*4+r
__global__ void prep_lift(const float* __restrict__ w1, f16* __restrict__ wmat) {
  int i = blockIdx.x * blockDim.x + threadIdx.x;
  if (i >= 32 * NPAD) return;
  int k = i / NPAD, n = i % NPAD;
  float v = 0.f;
  if (k < 9 && n < NCH) {
    int dy = k / 3, dx = k % 3;
    int c = n >> 2, r = n & 3;
    int sy, sx;
    switch (r) {
      case 0:  sy = dy;     sx = dx;     break;
      case 1:  sy = dx;     sx = 2 - dy; break;
      case 2:  sy = 2 - dy; sx = 2 - dx; break;
      default: sy = 2 - dx; sx = dy;     break;
    }
    v = w1[c * 9 + sy * 3 + sx];
  }
  wmat[i] = (f16)v;
}

// Regular->regular P4 kernel: full[co*4+r, ci*4+s, dy, dx] = rot90^r(w[co,ci,(s-r)&3])[dy,dx]
// Weight matrix: [Kpad][48], k = (dy*ks+dx)*40 + ci*4+s, n = co*4+r
__global__ void prep_gconv(const float* __restrict__ w, f16* __restrict__ wmat,
                           int ks, int Kpad) {
  int i = blockIdx.x * blockDim.x + threadIdx.x;
  if (i >= Kpad * NPAD) return;
  int k = i / NPAD, n = i % NPAD;
  int Kdim = ks * ks * NCH;
  float v = 0.f;
  if (k < Kdim && n < NCH) {
    int ciG = k % NCH;
    int dyx = k / NCH;
    int dy = dyx / ks, dx = dyx % ks;
    int ci = ciG >> 2, s = ciG & 3;
    int co = n >> 2,  r = n & 3;
    int s2 = (s - r) & 3;
    int e = ks - 1;
    int sy, sx;
    switch (r) {
      case 0:  sy = dy;     sx = dx;     break;
      case 1:  sy = dx;     sx = e - dy; break;
      case 2:  sy = e - dy; sx = e - dx; break;
      default: sy = e - dx; sx = dy;     break;
    }
    v = w[(((co * 10 + ci) * 4 + s2) * ks + sy) * ks + sx];
  }
  wmat[i] = (f16)v;
}

// ---------------- implicit-GEMM convolution with WMMA ----------------
// act:  NHWC f16 [B, HIN, WIN, CI]
// wmat: [Kpad][48] f16, k = (dy*KS+dx)*CI + ci
// out:  NHWC f16 [B, HO, WO, 40]
// Block = 128 threads = 4 waves. M-tile = 128 pixels; each wave owns 32 M
// rows as TWO 16-row A fragments that share the three 16-wide B fragments
// -> 6 WMMAs per K-chunk per wave per 2 barriers.
//
// LDS tiles are in "fragment order": position perm(k) maps
// K{0..7,16..23} -> 0..15 (lanes 0-15) and K{8..15,24..31} -> 16..31
// (lanes 16-31), so each lane's 16 fragment values are contiguous and
// load as one 32-byte vector (2x ds_load_b128).
__device__ __forceinline__ constexpr int kperm(int k) {
  return (k < 8) ? k : (k < 16) ? k + 8 : (k < 24) ? k - 8 : k;
}

template<int CI, int KS, int PAD, int HIN, int WIN, int HO, int WO>
__launch_bounds__(128)
__global__ void conv_wmma(const f16* __restrict__ act, const f16* __restrict__ wmat,
                          f16* __restrict__ out, int M_total) {
  constexpr int KDIM    = KS * KS * CI;
  constexpr int NCHUNKS = (KDIM + 31) / 32;
  // row stride 48 f16 = 96 B: every fragment half is 32-byte aligned
  __shared__ __attribute__((aligned(32))) f16 As[128][48];
  __shared__ __attribute__((aligned(32))) f16 Bt[48][48];   // [n][perm(k)]

  const int tid  = threadIdx.x;
  const int lane = tid & 31, wave = tid >> 5;
  const int lrow = lane & 15, lhi  = lane >> 4;
  const int m_base = blockIdx.x * 128;

  // each thread owns one full A row (all 32 K of a chunk) for the im2col fill
  const int m = m_base + tid;
  const bool mvalid = (m < M_total);
  int b = 0, y = 0, x = 0;
  if (mvalid) {
    b = m / (HO * WO);                 // compile-time divisors -> mul/shift
    int rem = m - b * (HO * WO);
    y = rem / WO;
    x = rem - y * WO;
  }
  const f16* __restrict__ actb = act + (long)b * (HIN * WIN * CI);

  v8f z = {0.f, 0.f, 0.f, 0.f, 0.f, 0.f, 0.f, 0.f};
  v8f acc[2][3];
  #pragma unroll
  for (int h = 0; h < 2; ++h) { acc[h][0] = z; acc[h][1] = z; acc[h][2] = z; }

  for (int kc = 0; kc < NCHUNKS; ++kc) {
    __syncthreads();
    // ---- im2col gather into registers, then 4x 16B LDS stores ----
    f16 vals[32];
    #pragma unroll
    for (int e = 0; e < 32; ++e) {
      int kk = kc * 32 + e;
      f16 v = (f16)0.f;
      if (mvalid && kk < KDIM) {
        int dyx = kk / CI;
        int ci  = kk - dyx * CI;
        int dy  = dyx / KS;
        int dx  = dyx - dy * KS;
        int iy = y + dy - PAD, ix = x + dx - PAD;
        if (iy >= 0 && iy < HIN && ix >= 0 && ix < WIN)
          v = actb[(iy * WIN + ix) * CI + ci];
      }
      vals[e] = v;
    }
    // perm placement: k0..7 -> pos0..7, k8..15 -> pos16..23,
    //                 k16..23 -> pos8..15, k24..31 -> pos24..31
    v8h g0, g1, g2, g3;
    #pragma unroll
    for (int j = 0; j < 8; ++j) {
      g0[j] = vals[j];      g1[j] = vals[8 + j];
      g2[j] = vals[16 + j]; g3[j] = vals[24 + j];
    }
    *(v8h*)&As[tid][0]  = g0;
    *(v8h*)&As[tid][16] = g1;
    *(v8h*)&As[tid][8]  = g2;
    *(v8h*)&As[tid][24] = g3;

    // ---- weight tile: contiguous reads, permuted-transposed LDS writes ----
    #pragma unroll
    for (int e = 0; e < 12; ++e) {
      int idx = tid * 12 + e;            // linear over [k][n] chunk
      int k = idx / 48, n = idx - k * 48;
      Bt[n][kperm(k)] = wmat[kc * (32 * 48) + idx];
    }
    // prefetch next weight chunk (global_prefetch_b8)
    {
      int kn = (kc + 1 < NCHUNKS) ? kc + 1 : kc;
      __builtin_prefetch(wmat + (long)kn * (32 * 48), 0, 1);
    }
    __syncthreads();

    // ---- fragments: one 32B vector load each ----
    v16h a0 = *(const v16h*)&As[wave * 32 + lrow][lhi * 16];
    v16h a1 = *(const v16h*)&As[wave * 32 + 16 + lrow][lhi * 16];
    v16h b0 = *(const v16h*)&Bt[lrow][lhi * 16];
    v16h b1 = *(const v16h*)&Bt[16 + lrow][lhi * 16];
    v16h b2 = *(const v16h*)&Bt[32 + lrow][lhi * 16];

    acc[0][0] = __builtin_amdgcn_wmma_f32_16x16x32_f16(false, a0, false, b0,
                                                       (short)0, acc[0][0], false, false);
    acc[0][1] = __builtin_amdgcn_wmma_f32_16x16x32_f16(false, a0, false, b1,
                                                       (short)0, acc[0][1], false, false);
    acc[0][2] = __builtin_amdgcn_wmma_f32_16x16x32_f16(false, a0, false, b2,
                                                       (short)0, acc[0][2], false, false);
    acc[1][0] = __builtin_amdgcn_wmma_f32_16x16x32_f16(false, a1, false, b0,
                                                       (short)0, acc[1][0], false, false);
    acc[1][1] = __builtin_amdgcn_wmma_f32_16x16x32_f16(false, a1, false, b1,
                                                       (short)0, acc[1][1], false, false);
    acc[1][2] = __builtin_amdgcn_wmma_f32_16x16x32_f16(false, a1, false, b2,
                                                       (short)0, acc[1][2], false, false);
  }

  // ---- epilogue: C/D layout VGPR r -> M = r + 8*lhi, N = lane&15 ----
  #pragma unroll
  for (int h = 0; h < 2; ++h) {
    #pragma unroll
    for (int nt = 0; nt < 3; ++nt) {
      int co = nt * 16 + lrow;
      if (co < NCH) {
        #pragma unroll
        for (int r = 0; r < 8; ++r) {
          int mm = m_base + wave * 32 + h * 16 + r + lhi * 8;
          if (mm < M_total) out[(long)mm * NCH + co] = (f16)acc[h][nt][r];
        }
      }
    }
  }
}

// ---------------- InnerBatchNorm (stats over B,G,H,W per field) ----------------
// blockDim=320 and grid stride are multiples of 40 -> each thread's channel is fixed.
__launch_bounds__(320)
__global__ void bn_stats(const f16* __restrict__ act, float* __restrict__ stats, long total) {
  __shared__ float ssum[10], ssq[10];
  int tid = threadIdx.x;
  if (tid < 10) { ssum[tid] = 0.f; ssq[tid] = 0.f; }
  __syncthreads();
  long i0 = (long)blockIdx.x * 320 + tid;
  long stride = (long)gridDim.x * 320;
  int f = ((int)(i0 % NCH)) >> 2;     // constant for this thread
  float s = 0.f, q = 0.f;
  for (long i = i0; i < total; i += stride) {
    float v = (float)act[i];
    s += v; q += v * v;
  }
  atomicAdd(&ssum[f], s);
  atomicAdd(&ssq[f], q);
  __syncthreads();
  if (tid < 10) {
    atomicAdd(&stats[tid], ssum[tid]);
    atomicAdd(&stats[10 + tid], ssq[tid]);
  }
}

__global__ void bn_relu(f16* __restrict__ act, const float* __restrict__ stats,
                        const float* __restrict__ gamma, const float* __restrict__ beta,
                        long total, float invcnt) {
  __shared__ float sc[10], sh[10];
  int tid = threadIdx.x;
  if (tid < 10) {
    float mean = stats[tid] * invcnt;
    float var  = stats[10 + tid] * invcnt - mean * mean;
    float scale = gamma[tid] * rsqrtf(var + EPSV);
    sc[tid] = scale;
    sh[tid] = beta[tid] - mean * scale;
  }
  __syncthreads();
  long stride = (long)gridDim.x * blockDim.x;
  for (long i = (long)blockIdx.x * blockDim.x + tid; i < total; i += stride) {
    int f = ((int)(i % NCH)) >> 2;
    float v = (float)act[i] * sc[f] + sh[f];
    act[i] = (f16)fmaxf(v, 0.f);
  }
}

// ---------------- 2x2 maxpool, NHWC 28->14 ----------------
__global__ void maxpool2(const f16* __restrict__ in, f16* __restrict__ out, int B) {
  long total = (long)B * 14 * 14 * NCH;
  long i = (long)blockIdx.x * blockDim.x + threadIdx.x;
  if (i >= total) return;
  int c = (int)(i % NCH);
  long t = i / NCH;
  int x = (int)(t % 14); t /= 14;
  int y = (int)(t % 14); t /= 14;
  int b = (int)t;
  const f16* p = in + (((long)b * 28 + 2 * y) * 28 + 2 * x) * NCH + c;
  float v0 = (float)p[0], v1 = (float)p[NCH];
  float v2 = (float)p[28 * NCH], v3 = (float)p[28 * NCH + NCH];
  out[i] = (f16)fmaxf(fmaxf(v0, v1), fmaxf(v2, v3));
}

// ---------------- group max-pool over orientations + flatten (C,11,11) ----------------
__global__ void group_pool(const f16* __restrict__ act, float* __restrict__ pooled, int B) {
  long total = (long)B * 10 * 121;
  long i = (long)blockIdx.x * blockDim.x + threadIdx.x;
  if (i >= total) return;
  int x = (int)(i % 11); long t = i / 11;
  int y = (int)(t % 11); t /= 11;
  int c = (int)(t % 10); t /= 10;
  int b = (int)t;
  const f16* p = act + (((long)b * 11 + y) * 11 + x) * NCH + c * 4;
  float m = (float)p[0];
  m = fmaxf(m, (float)p[1]);
  m = fmaxf(m, (float)p[2]);
  m = fmaxf(m, (float)p[3]);
  pooled[(long)b * 1210 + c * 121 + y * 11 + x] = m;
}

// ---------------- final FC [B,1210] @ [10,1210]^T + bias ----------------
__global__ void fc_kernel(const float* __restrict__ pooled, const float* __restrict__ wfc,
                          const float* __restrict__ bfc, float* __restrict__ out, int B) {
  int i = blockIdx.x * blockDim.x + threadIdx.x;
  if (i >= B * 10) return;
  int b = i / 10, j = i % 10;
  const float* h = pooled + (long)b * 1210;
  const float* w = wfc + (long)j * 1210;
  float acc = bfc[j];
  for (int t = 0; t < 1210; ++t) acc += h[t] * w[t];
  out[i] = acc;
}

// ---------------- host orchestration ----------------
extern "C" void kernel_launch(void* const* d_in, const int* in_sizes, int n_in,
                              void* d_out, int out_size, void* d_ws, size_t ws_size,
                              hipStream_t stream) {
  (void)in_sizes; (void)n_in; (void)out_size; (void)ws_size;
  const float* x   = (const float*)d_in[0];
  const float* w1  = (const float*)d_in[1];
  const float* w2  = (const float*)d_in[2];
  const float* w3  = (const float*)d_in[3];
  const float* w4  = (const float*)d_in[4];
  const float* w5  = (const float*)d_in[5];
  const float* w6  = (const float*)d_in[6];
  const float* w7  = (const float*)d_in[7];
  const float* g1  = (const float*)d_in[8];
  const float* b1  = (const float*)d_in[9];
  const float* g2  = (const float*)d_in[10];
  const float* b2  = (const float*)d_in[11];
  const float* gs  = (const float*)d_in[12];
  const float* bs  = (const float*)d_in[13];
  const float* wfc = (const float*)d_in[14];
  const float* bfc = (const float*)d_in[15];
  float* out = (float*)d_out;

  const int B = 1024;
  char* ws = (char*)d_ws;
  size_t off = 0;
  auto alloc = [&](size_t bytes) -> char* {
    char* p = ws + off;
    off += (bytes + 255) & ~(size_t)255;
    return p;
  };
  f16* x16  = (f16*)alloc((size_t)B * 28 * 28 * sizeof(f16));
  f16* bufA = (f16*)alloc((size_t)B * 28 * 28 * NCH * sizeof(f16));
  f16* bufB = (f16*)alloc((size_t)B * 28 * 28 * NCH * sizeof(f16));
  f16* wm1  = (f16*)alloc((size_t)32  * NPAD * sizeof(f16));
  f16* wm2  = (f16*)alloc((size_t)384 * NPAD * sizeof(f16));
  f16* wm3  = (f16*)alloc((size_t)384 * NPAD * sizeof(f16));
  f16* wm4  = (f16*)alloc((size_t)384 * NPAD * sizeof(f16));
  f16* wm5  = (f16*)alloc((size_t)384 * NPAD * sizeof(f16));
  f16* wm6  = (f16*)alloc((size_t)384 * NPAD * sizeof(f16));
  f16* wm7  = (f16*)alloc((size_t)640 * NPAD * sizeof(f16));
  float* stats  = (float*)alloc(7 * 20 * sizeof(float));
  float* pooled = (float*)alloc((size_t)B * 1210 * sizeof(float));

  // zero per-call stats accumulators (graph-replay safe)
  zero_f32<<<1, 256, 0, stream>>>(stats, 140);

  // input f32 NCHW (C=1) == NHWC -> f16
  {
    long n = (long)B * 784;
    cvt_f32_f16<<<(unsigned)((n + 255) / 256), 256, 0, stream>>>(x, x16, n);
  }
  // expand group-equivariant kernels into padded [Kpad x 48] f16 matrices
  prep_lift <<<(32  * NPAD + 127) / 128, 128, 0, stream>>>(w1, wm1);
  prep_gconv<<<(384 * NPAD + 127) / 128, 128, 0, stream>>>(w2, wm2, 3, 384);
  prep_gconv<<<(384 * NPAD + 127) / 128, 128, 0, stream>>>(w3, wm3, 3, 384);
  prep_gconv<<<(384 * NPAD + 127) / 128, 128, 0, stream>>>(w4, wm4, 3, 384);
  prep_gconv<<<(384 * NPAD + 127) / 128, 128, 0, stream>>>(w5, wm5, 3, 384);
  prep_gconv<<<(384 * NPAD + 127) / 128, 128, 0, stream>>>(w6, wm6, 3, 384);
  prep_gconv<<<(640 * NPAD + 127) / 128, 128, 0, stream>>>(w7, wm7, 4, 640);

  const int M1 = B * 28 * 28;
  const int M2 = B * 14 * 14;
  const int M3 = B * 11 * 11;
  const long tot1 = (long)M1 * NCH, tot2 = (long)M2 * NCH, tot3 = (long)M3 * NCH;
  const unsigned gb1 = (unsigned)((M1 + 127) / 128);
  const unsigned gb2 = (unsigned)((M2 + 127) / 128);
  const unsigned gb3 = (unsigned)((M3 + 127) / 128);

  // layer 1: lift conv 3x3 pad1, Ci=1 (K=9 -> 1 chunk)
  conv_wmma<1, 3, 1, 28, 28, 28, 28><<<gb1, 128, 0, stream>>>(x16, wm1, bufA, M1);
  bn_stats<<<512, 320, 0, stream>>>(bufA, stats + 0, tot1);
  bn_relu <<<1024, 256, 0, stream>>>(bufA, stats + 0, g1, b1, tot1, 1.f / (float)((long)M1 * 4));

  // layer 2: gconv 3x3 pad1 (K=360 -> 12 chunks), then BN+ReLU, then maxpool
  conv_wmma<40, 3, 1, 28, 28, 28, 28><<<gb1, 128, 0, stream>>>(bufA, wm2, bufB, M1);
  bn_stats<<<512, 320, 0, stream>>>(bufB, stats + 20, tot1);
  bn_relu <<<1024, 256, 0, stream>>>(bufB, stats + 20, g2, b2, tot1, 1.f / (float)((long)M1 * 4));
  {
    long tp = (long)B * 14 * 14 * NCH;
    maxpool2<<<(unsigned)((tp + 255) / 256), 256, 0, stream>>>(bufB, bufA, B);
  }

  // layers 3-6: gconv 3x3 pad1 on 14x14
  conv_wmma<40, 3, 1, 14, 14, 14, 14><<<gb2, 128, 0, stream>>>(bufA, wm3, bufB, M2);
  bn_stats<<<512, 320, 0, stream>>>(bufB, stats + 40, tot2);
  bn_relu <<<512, 256, 0, stream>>>(bufB, stats + 40, gs, bs, tot2, 1.f / (float)((long)M2 * 4));

  conv_wmma<40, 3, 1, 14, 14, 14, 14><<<gb2, 128, 0, stream>>>(bufB, wm4, bufA, M2);
  bn_stats<<<512, 320, 0, stream>>>(bufA, stats + 60, tot2);
  bn_relu <<<512, 256, 0, stream>>>(bufA, stats + 60, gs, bs, tot2, 1.f / (float)((long)M2 * 4));

  conv_wmma<40, 3, 1, 14, 14, 14, 14><<<gb2, 128, 0, stream>>>(bufA, wm5, bufB, M2);
  bn_stats<<<512, 320, 0, stream>>>(bufB, stats + 80, tot2);
  bn_relu <<<512, 256, 0, stream>>>(bufB, stats + 80, gs, bs, tot2, 1.f / (float)((long)M2 * 4));

  conv_wmma<40, 3, 1, 14, 14, 14, 14><<<gb2, 128, 0, stream>>>(bufB, wm6, bufA, M2);
  bn_stats<<<512, 320, 0, stream>>>(bufA, stats + 100, tot2);
  bn_relu <<<512, 256, 0, stream>>>(bufA, stats + 100, gs, bs, tot2, 1.f / (float)((long)M2 * 4));

  // layer 7: gconv 4x4 pad0, 14 -> 11 (K=640 -> 20 chunks)
  conv_wmma<40, 4, 0, 14, 14, 11, 11><<<gb3, 128, 0, stream>>>(bufA, wm7, bufB, M3);
  bn_stats<<<512, 320, 0, stream>>>(bufB, stats + 120, tot3);
  bn_relu <<<512, 256, 0, stream>>>(bufB, stats + 120, gs, bs, tot3, 1.f / (float)((long)M3 * 4));

  // orientation max-pool + flatten, then FC
  {
    long gp = (long)B * 1210;
    group_pool<<<(unsigned)((gp + 255) / 256), 256, 0, stream>>>(bufB, pooled, B);
  }
  fc_kernel<<<(B * 10 + 255) / 256, 256, 0, stream>>>(pooled, wfc, bfc, out, B);
}
